// BlockLBPUNet_51642686767174
// MI455X (gfx1250) — matrix operand order repaired and checked by reading, not compiled, not run on hardware
//
#include <hip/hip_runtime.h>
#include <hip/hip_bf16.h>

typedef __attribute__((ext_vector_type(16))) __bf16 v16bf;
typedef __attribute__((ext_vector_type(8)))  __bf16 v8bf;
typedef __attribute__((ext_vector_type(8)))  float  v8f;

#define CP 136                      // padded channel stride (16B aligned, bank-spread)
#define XN_ELEMS (3*66*CP)          // staged normalized input tile
#define Y_OFF    XN_ELEMS           // y (post-conv) tile region

__device__ __forceinline__ v16bf ldfrag(const __bf16* p) {
    v8bf lo = *(const v8bf*)p;
    v8bf hi = *(const v8bf*)(p + 8);
    v16bf r;
    #pragma unroll
    for (int i = 0; i < 8; ++i) { r[i] = lo[i]; r[i + 8] = hi[i]; }
    return r;
}

// -------- Kernel 1: per-channel batch statistics -> affine scale/shift --------
__global__ __launch_bounds__(256)
void bn_stats_kernel(const float* __restrict__ x, const float* __restrict__ gamma,
                     const float* __restrict__ beta, float* __restrict__ scale,
                     float* __restrict__ shift)
{
    const int c = blockIdx.x;        // 128 channels
    const int tid = threadIdx.x;
    float s = 0.f, sq = 0.f;
    for (int b = 0; b < 8; ++b) {
        const float* p = x + ((size_t)(b*128 + c)) * 65536;
        for (int i = tid; i < 65536; i += 256) {
            float v = p[i];
            s += v; sq += v * v;
        }
    }
    __shared__ float rs[256], rq[256];
    rs[tid] = s; rq[tid] = sq;
    __syncthreads();
    for (int o = 128; o > 0; o >>= 1) {
        if (tid < o) { rs[tid] += rs[tid+o]; rq[tid] += rq[tid+o]; }
        __syncthreads();
    }
    if (tid == 0) {
        const float n = 524288.f;
        float mean = rs[0] / n;
        float var  = rq[0] / n - mean * mean;
        float a = gamma[c] * rsqrtf(var + 1e-5f);
        scale[c] = a;
        shift[c] = beta[c] - mean * a;
    }
}

// -------- Kernel 2: pack ternary 3x3 weights into WMMA A-fragment layout --------
// K ordering: K = tap*128 + c ; kstep s covers c = (s&3)*32 .. +31 of tap = s>>2
// A-fragment k order used here (contiguous, matches B): lane group g=lane/16 ; value v: k_local = v + g*16
__global__ __launch_bounds__(256)
void pack_lbp_kernel(const float* __restrict__ w, __bf16* __restrict__ WA)
{
    int idx = blockIdx.x * 256 + threadIdx.x;      // < 8*36*32*16 = 147456
    int v    = idx & 15;
    int lane = (idx >> 4) & 31;
    int rest = idx >> 9;
    int s    = rest % 36;
    int mt   = rest / 36;
    int g = lane >> 4, m = lane & 15;
    int cout = mt * 16 + m;
    int kl = v + g * 16;                           // 0..31
    int tap = s >> 2;
    int c   = ((s & 3) << 5) + kl;
    int ky = tap / 3, kx = tap - ky * 3;
    WA[idx] = (__bf16)w[((cout * 128 + c) * 3 + ky) * 3 + kx];
}

// -------- Kernel 3: pack 1x1 weights into WMMA A-fragment layout --------
__global__ __launch_bounds__(256)
void pack_w1_kernel(const float* __restrict__ w, __bf16* __restrict__ W1A)
{
    int idx = blockIdx.x * 256 + threadIdx.x;      // < 8*4*32*16 = 16384
    int v    = idx & 15;
    int lane = (idx >> 4) & 31;
    int rest = idx >> 9;
    int s    = rest & 3;
    int mt   = rest >> 2;
    int g = lane >> 4, m = lane & 15;
    int cout = mt * 16 + m;
    int kl = v + g * 16;
    int c  = (s << 5) + kl;
    W1A[idx] = (__bf16)w[cout * 128 + c];
}

// -------- Kernel 4: fused normalize -> 3x3 conv (WMMA) -> relu+res -> 1x1 (WMMA) --------
// Wave tiling: wave w owns mtiles {2*(w&3), 2*(w&3)+1} and a 32-col half (w>>2).
// Each B fragment feeds 2 WMMAs and each A fragment feeds 2 WMMAs (halved LDS traffic).
__global__ __launch_bounds__(256)
void lbp_main_kernel(const float* __restrict__ x, const float* __restrict__ scale,
                     const float* __restrict__ shift, const __bf16* __restrict__ WA,
                     const __bf16* __restrict__ W1A, const float* __restrict__ bias,
                     float* __restrict__ out)
{
    __shared__ __bf16 smem[XN_ELEMS + 64 * CP];
    __shared__ float s_sc[128], s_sh[128], s_bi[128];

    const int tid  = threadIdx.x;
    const int lane = tid & 31;
    const int wv   = tid >> 5;          // wave id 0..7
    const int g    = lane >> 4;         // lane half
    const int ln   = lane & 15;         // column within N-tile
    const int w0   = blockIdx.x * 64;
    const int h    = blockIdx.y;
    const int b    = blockIdx.z;

    const int mt0 = (wv & 3) * 2;       // first of this wave's two channel tiles
    const int ntb = (wv >> 2) * 32;     // column base of this wave's two N-tiles

    if (tid < 128) {
        s_sc[tid] = scale[tid];
        s_sh[tid] = shift[tid];
        s_bi[tid] = bias[tid];
    }
    __syncthreads();

    // ---- Stage xn = a_c*x + b_c (zero at conv padding) as bf16, layout [row][col][ch] ----
    // Thread owns a fixed (row,col) site, loops channels (contiguous in LDS -> b128 stores).
    {
        const int stcol = tid % 66;
        const int strow = tid / 66;      // 0..3; 198 active threads
        if (strow < 3) {
            const int hh = h - 1 + strow;
            const int ww = w0 - 1 + stcol;
            const bool inb = (hh >= 0 && hh < 256 && ww >= 0 && ww < 256);
            __bf16* dst = smem + (strow * 66 + stcol) * CP;
            const float* src = x + (size_t)b * 128 * 65536 + (size_t)hh * 256 + ww;
            for (int c = 0; c < 128; c += 8) {
                v8bf pk;
                #pragma unroll
                for (int u = 0; u < 8; ++u) {
                    float v = 0.f;
                    if (inb) v = fmaf(src[(size_t)(c + u) * 65536], s_sc[c + u], s_sh[c + u]);
                    pk[u] = (__bf16)v;
                }
                *(v8bf*)(dst + c) = pk;
            }
        }
    }
    __syncthreads();

    // ---- 3x3 conv as GEMM: D[32 x 32] += Wlbp[32 x 1152] * xn_im2col[1152 x 32] ----
    v8f acc[4] = {};    // [mt(2)][nt(2)]
    for (int s = 0; s < 36; ++s) {
        v16bf a0 = ldfrag(WA + (((size_t)( mt0      * 36 + s) * 32 + lane) << 4));
        v16bf a1 = ldfrag(WA + (((size_t)((mt0 + 1) * 36 + s) * 32 + lane) << 4));

        int tap = s >> 2;
        int ky = tap / 3, kx = tap - ky * 3;
        int cb = ((s & 3) << 5) + (g << 4);       // 16 consecutive channels per lane
        const __bf16* pb = smem + (ky * 66 + ntb + ln + kx) * CP + cb;
        v16bf b0 = ldfrag(pb);
        v16bf b1 = ldfrag(pb + 16 * CP);

        acc[0] = __builtin_amdgcn_wmma_f32_16x16x32_bf16(false, a0, false, b0, (short)0, acc[0], false, false);
        acc[1] = __builtin_amdgcn_wmma_f32_16x16x32_bf16(false, a0, false, b1, (short)0, acc[1], false, false);
        acc[2] = __builtin_amdgcn_wmma_f32_16x16x32_bf16(false, a1, false, b0, (short)0, acc[2], false, false);
        acc[3] = __builtin_amdgcn_wmma_f32_16x16x32_bf16(false, a1, false, b1, (short)0, acc[3], false, false);
    }

    // ---- relu + residual, write y as bf16 to LDS [col][ch] ----
    #pragma unroll
    for (int i = 0; i < 2; ++i) {
        const int chb = (mt0 + i) * 16 + g * 8;   // 8 consecutive channels per lane
        #pragma unroll
        for (int j = 0; j < 2; ++j) {
            const int colw = ntb + j * 16 + ln;
            const int w = w0 + colw;
            v8bf yv;
            #pragma unroll
            for (int r = 0; r < 8; ++r) {
                float xr = x[((size_t)(b * 128 + chb + r)) * 65536 + (size_t)h * 256 + w];
                float v = acc[i * 2 + j][r];
                v = v > 0.f ? v : 0.f;
                yv[r] = (__bf16)(v + xr);
            }
            *(v8bf*)(smem + Y_OFF + colw * CP + chb) = yv;
        }
    }
    __syncthreads();

    // ---- 1x1 conv as GEMM: out[32 x 32] += W1[32 x 128] * y[128 x 32] ----
    v8f acc2[4] = {};
    #pragma unroll
    for (int s2 = 0; s2 < 4; ++s2) {
        v16bf a0 = ldfrag(W1A + (((size_t)( mt0      * 4 + s2) * 32 + lane) << 4));
        v16bf a1 = ldfrag(W1A + (((size_t)((mt0 + 1) * 4 + s2) * 32 + lane) << 4));
        int cb = (s2 << 5) + (g << 4);
        const __bf16* pb = smem + Y_OFF + (ntb + ln) * CP + cb;
        v16bf b0 = ldfrag(pb);
        v16bf b1 = ldfrag(pb + 16 * CP);

        acc2[0] = __builtin_amdgcn_wmma_f32_16x16x32_bf16(false, a0, false, b0, (short)0, acc2[0], false, false);
        acc2[1] = __builtin_amdgcn_wmma_f32_16x16x32_bf16(false, a0, false, b1, (short)0, acc2[1], false, false);
        acc2[2] = __builtin_amdgcn_wmma_f32_16x16x32_bf16(false, a1, false, b0, (short)0, acc2[2], false, false);
        acc2[3] = __builtin_amdgcn_wmma_f32_16x16x32_bf16(false, a1, false, b1, (short)0, acc2[3], false, false);
    }

    // ---- bias + store f32 output ----
    #pragma unroll
    for (int i = 0; i < 2; ++i) {
        const int chb = (mt0 + i) * 16 + g * 8;
        #pragma unroll
        for (int j = 0; j < 2; ++j) {
            const int w = w0 + ntb + j * 16 + ln;
            #pragma unroll
            for (int r = 0; r < 8; ++r) {
                const int o = chb + r;
                out[((size_t)(b * 128 + o)) * 65536 + (size_t)h * 256 + w] = acc2[i * 2 + j][r] + s_bi[o];
            }
        }
    }
}

extern "C" void kernel_launch(void* const* d_in, const int* in_sizes, int n_in,
                              void* d_out, int out_size, void* d_ws, size_t ws_size,
                              hipStream_t stream) {
    const float* x     = (const float*)d_in[0];   // (8,128,256,256)
    const float* gamma = (const float*)d_in[1];   // (128)
    const float* beta  = (const float*)d_in[2];   // (128)
    const float* wlbp  = (const float*)d_in[3];   // (128,128,3,3)
    const float* w1    = (const float*)d_in[4];   // (128,128,1,1)
    const float* b1    = (const float*)d_in[5];   // (128)
    float* out = (float*)d_out;

    // workspace layout
    float*  scale = (float*)d_ws;                              // 128 f32
    float*  shift = scale + 128;                               // 128 f32
    __bf16* WA    = (__bf16*)((char*)d_ws + 1024);             // 147456 bf16
    __bf16* W1A   = WA + 147456;                               // 16384 bf16

    bn_stats_kernel<<<128, 256, 0, stream>>>(x, gamma, beta, scale, shift);
    pack_lbp_kernel<<<576, 256, 0, stream>>>(wlbp, WA);
    pack_w1_kernel<<<64, 256, 0, stream>>>(w1, W1A);
    lbp_main_kernel<<<dim3(4, 256, 8), 256, 0, stream>>>(x, scale, shift, WA, W1A, b1, out);
}
